// RNNDefine_52140902973488
// MI455X (gfx1250) — matrix operand order repaired and checked by reading, not compiled, not run on hardware
//
#include <hip/hip_runtime.h>
#include <hip/hip_bf16.h>
#include <stdint.h>

// ---------------------------------------------------------------------------
// tanh-RNN for MI455X (gfx1250, wave32, WMMA bf16 16x16x32, f32 accumulate)
//   xproj = x @ Wx + bx                 : compute-bound GEMM, LDS-staged B
//   h_t   = tanh(xproj_t + h @ Wh + bh) : 256 dependent steps, Wh L2-resident
// B (weights) packed once into WMMA-B-fragment tiles; A (x, h) stays
// row-major bf16 (A-fragments = two contiguous b128 loads per lane).
// xproj GEMM stages its 128KB B panel into LDS via async global->LDS copies
// (GLOBAL_LOAD_ASYNC_TO_LDS_B128, ASYNCcnt / s_wait_asynccnt) so all 8 waves
// of a workgroup share a single L2 read of the panel.
// ---------------------------------------------------------------------------

#define SEQ    256
#define BATCH  64
#define INPUT  1024
#define HIDDEN 2048
#define MROWS  (SEQ * BATCH)      // 16384

typedef __attribute__((ext_vector_type(16))) __bf16        v16bf;
typedef __attribute__((ext_vector_type(8)))  float         v8f;
typedef __attribute__((ext_vector_type(4)))  unsigned int  v4u;

union FragU { v16bf v; v4u u[2]; };

#if __has_builtin(__builtin_amdgcn_global_load_async_to_lds_b128)
#define HAS_ASYNC_LDS 1
// Builtin expects: (v4i addrspace(1)*, v4i addrspace(3)*, imm int, imm int)
// (clang prints AS1 as "__device__" in HIP mode — see round-2 diagnostic).
typedef int v4i_t __attribute__((vector_size(16)));
typedef __attribute__((address_space(1))) v4i_t as1_v4i;
typedef __attribute__((address_space(3))) v4i_t as3_v4i;
#else
#define HAS_ASYNC_LDS 0
#endif

// ----- A fragment (16x32 bf16) from ROW-MAJOR bf16 matrix ------------------
// lane l: row = rbase + (l&15); i=0..7  -> k = kbase + (l>>4)*8 + i
//                               i=8..15 -> k = kbase + 16 + (l>>4)*8 + (i-8)
__device__ __forceinline__ v16bf load_A_frag(const __bf16* __restrict__ A,
                                             int rbase, int kbase, int ld, int lane) {
    const int m    = rbase + (lane & 15);
    const int half = lane >> 4;
    const __bf16* p = A + (size_t)m * ld + kbase + half * 8;
    FragU f;
    f.u[0] = *(const v4u*)(p);        // k = kbase+half*8    .. +7
    f.u[1] = *(const v4u*)(p + 16);   // k = kbase+16+half*8 .. +7
    return f.v;
}

// ----- B fragment (32x16 bf16) from PACKED weights (global) ----------------
// tile (kt,nt): 512 bf16 = 1KB; lane l owns 16 contiguous bf16 at l*16.
// Packed element (tile,lane,i) == W[kt*32 + (lane>>4)*8 + (i&7) + (i>>3)*16][nt*16 + (lane&15)]
__device__ __forceinline__ v16bf load_B_frag(const __bf16* __restrict__ BP,
                                             int kt, int nt, int ntiles, int lane) {
    const __bf16* p = BP + ((size_t)(kt * ntiles + nt) << 9) + lane * 16;
    FragU f;
    f.u[0] = *(const v4u*)(p);
    f.u[1] = *(const v4u*)(p + 8);
    return f.v;
}

// ----- B fragment from an LDS-staged panel (tile-linear) -------------------
__device__ __forceinline__ v16bf load_B_frag_lds(const __bf16* Bs, int tile, int lane) {
    const __bf16* p = Bs + ((size_t)tile << 9) + lane * 16;
    FragU f;
    f.u[0] = *(const v4u*)(p);      // -> ds_load_b128
    f.u[1] = *(const v4u*)(p + 8);
    return f.v;
}

// ----- stage a K x 64-col packed-B panel into LDS --------------------------
// Panel = kts k-tiles x 4 n-tiles, tile-linear in LDS: (kt*4 + j)*1KB.
// Global packed layout has the 4 n-tiles of one kt contiguous (4KB chunk).
__device__ __forceinline__ void stage_B_panel(const __bf16* __restrict__ BPg,
                                              __bf16* Bs, int nt0, int ntiles, int kts) {
    const int total16 = kts * 256;                   // # of 16-byte chunks
    for (int c = threadIdx.x; c < total16; c += 256) {
        const int o  = c << 4;                       // byte offset in panel
        const int kt = o >> 12;                      // 4KB per kt
        const int w  = o & 4095;
        const char* src = (const char*)BPg + (((size_t)(kt * ntiles + nt0)) << 10) + w;
        char*       dst = (char*)Bs + o;
#if HAS_ASYNC_LDS
        __builtin_amdgcn_global_load_async_to_lds_b128(
            (as1_v4i*)(uintptr_t)src,
            (as3_v4i*)(uint32_t)(uintptr_t)dst, 0, 0);
#else
        *(v4u*)dst = *(const v4u*)src;
#endif
    }
#if HAS_ASYNC_LDS
#if __has_builtin(__builtin_amdgcn_s_wait_asynccnt)
    __builtin_amdgcn_s_wait_asynccnt(0);
#else
    asm volatile("s_wait_asynccnt 0x0" ::: "memory");
#endif
#endif
    __syncthreads();
}

// ----- prep: f32 -> bf16 elementwise ---------------------------------------
__global__ void cvt_f32_bf16_kernel(const float* __restrict__ src,
                                    __bf16* __restrict__ dst, size_t n) {
    for (size_t i = (size_t)blockIdx.x * blockDim.x + threadIdx.x; i < n;
         i += (size_t)gridDim.x * blockDim.x)
        dst[i] = (__bf16)src[i];
}

// ----- prep: pack row-major f32 (K,N) weights -> bf16 B-fragment tiles -----
__global__ void pack_B_kernel(const float* __restrict__ src,
                              __bf16* __restrict__ dst, int K, int N) {
    const int ntiles = N >> 4;
    const size_t total = (size_t)K * N;
    for (size_t e = (size_t)blockIdx.x * blockDim.x + threadIdx.x; e < total;
         e += (size_t)gridDim.x * blockDim.x) {
        const size_t tile = e >> 9;         // /512
        const int within  = (int)(e & 511);
        const int lane    = within >> 4;    // 0..31
        const int i       = within & 15;
        const int kt      = (int)(tile / ntiles);
        const int nt      = (int)(tile % ntiles);
        const int n = (nt << 4) + (lane & 15);
        const int k = (kt << 5) + ((lane >> 4) << 3) + (i & 7) + ((i >> 3) << 4);
        dst[e] = (__bf16)src[(size_t)k * N + n];
    }
}

// ----- xproj GEMM: out[m,n] = sum_k Xb[m,k]*Wx[k,n] + bx[n] ----------------
// grid (HIDDEN/64, MROWS/256), block 256 (8 waves). WG: 256 rows x 64 cols.
// B panel (1024x64 bf16 = 128KB) staged in LDS once, shared by all 8 waves.
// Wave w: rows [blockIdx.y*256 + w*32, +32), cols [blockIdx.x*64, +64)
//   => 2 m-tiles x 4 n-tiles = 8 WMMAs per 32-wide K step.
__global__ void __launch_bounds__(256)
gemm_xproj_kernel(const __bf16* __restrict__ Xb,
                  const __bf16* __restrict__ WxP,
                  const float*  __restrict__ bx,
                  float* __restrict__ out) {
    extern __shared__ __attribute__((aligned(16))) char smem[];
    __bf16* Bs = (__bf16*)smem;                       // 128KB panel

    const int lane = threadIdx.x & 31;
    const int wave = threadIdx.x >> 5;
    const int n0   = blockIdx.x * 64;

    stage_B_panel(WxP, Bs, n0 >> 4, HIDDEN >> 4, INPUT / 32);

    const int row = blockIdx.y * 256 + wave * 32;
    const int nl  = lane & 15, mh = lane >> 4;

    v8f c[2][4];
#pragma unroll
    for (int j = 0; j < 4; ++j) {
        const float b = bx[n0 + j * 16 + nl];
#pragma unroll
        for (int v = 0; v < 8; ++v) { c[0][j][v] = b; c[1][j][v] = b; }
    }

    for (int k = 0; k < INPUT; k += 32) {
        const v16bf a0 = load_A_frag(Xb, row,      k, INPUT, lane);
        const v16bf a1 = load_A_frag(Xb, row + 16, k, INPUT, lane);
#pragma unroll
        for (int j = 0; j < 4; ++j) {
            const v16bf b = load_B_frag_lds(Bs, (k >> 5) * 4 + j, lane);
            c[0][j] = __builtin_amdgcn_wmma_f32_16x16x32_bf16(
                false, a0, false, b, (short)0, c[0][j], false, false);
            c[1][j] = __builtin_amdgcn_wmma_f32_16x16x32_bf16(
                false, a1, false, b, (short)0, c[1][j], false, false);
        }
    }

#pragma unroll
    for (int mt = 0; mt < 2; ++mt) {
#pragma unroll
        for (int j = 0; j < 4; ++j) {
            const int n = n0 + j * 16 + nl;
#pragma unroll
            for (int v = 0; v < 8; ++v) {
                const int m = row + mt * 16 + mh * 8 + v;
                out[(size_t)m * HIDDEN + n] = c[mt][j][v];
            }
        }
    }
}

// ----- one recurrence step: h_new = tanh(xproj_t + h @ Wh + bh) ------------
// grid HIDDEN/128 = 16 blocks, 256 threads (8 waves).
// Wave = 64 rows x 16 cols (all 4 m-tiles, one n-tile): each Wh element is
// loaded exactly once per step (8MB/step, L2-resident); one B fragment feeds
// 4 WMMAs. h (A) is row-major bf16, 256KB working set.
__global__ void __launch_bounds__(256)
rnn_step_kernel(const __bf16* __restrict__ h_cur,
                const __bf16* __restrict__ WhP,
                const float*  __restrict__ bh,
                float*  __restrict__ out_t,   // xproj in, h out (fp32)
                __bf16* __restrict__ h_next) {
    const int lane  = threadIdx.x & 31;
    const int wave  = threadIdx.x >> 5;
    const int ntile = blockIdx.x * 8 + wave;      // 0..127
    const int n0    = ntile * 16;
    const int nl    = lane & 15, mh = lane >> 4;

    v8f c[4];
    {
        const float b = bh[n0 + nl];
#pragma unroll
        for (int mt = 0; mt < 4; ++mt)
#pragma unroll
            for (int v = 0; v < 8; ++v) c[mt][v] = b;
    }

    for (int k = 0; k < HIDDEN; k += 32) {
        const v16bf b = load_B_frag(WhP, k >> 5, ntile, HIDDEN >> 4, lane);
#pragma unroll
        for (int mt = 0; mt < 4; ++mt) {
            const v16bf a = load_A_frag(h_cur, mt * 16, k, HIDDEN, lane);
            c[mt] = __builtin_amdgcn_wmma_f32_16x16x32_bf16(
                false, a, false, b, (short)0, c[mt], false, false);
        }
    }

#pragma unroll
    for (int mt = 0; mt < 4; ++mt) {
#pragma unroll
        for (int v = 0; v < 8; ++v) {
            const int m = mt * 16 + mh * 8 + v;
            const size_t idx = (size_t)m * HIDDEN + n0 + nl;
            const float val = tanhf(c[mt][v] + out_t[idx]);
            out_t[idx]  = val;            // outputs[t] (fp32)
            h_next[idx] = (__bf16)val;    // recurrent state (bf16)
        }
    }
}

extern "C" void kernel_launch(void* const* d_in, const int* in_sizes, int n_in,
                              void* d_out, int out_size, void* d_ws, size_t ws_size,
                              hipStream_t stream) {
    const float* x  = (const float*)d_in[0];   // (256,64,1024)
    const float* h0 = (const float*)d_in[1];   // (1,64,2048)
    const float* Wx = (const float*)d_in[2];   // (1024,2048)
    const float* bx = (const float*)d_in[3];   // (2048,)
    const float* Wh = (const float*)d_in[4];   // (2048,2048)
    const float* bh = (const float*)d_in[5];   // (2048,)
    float* out = (float*)d_out;                // outputs (256,64,2048) ++ hn (64,2048)

    // workspace layout
    char* ws = (char*)d_ws;
    size_t off = 0;
    __bf16* Xb  = (__bf16*)(ws + off); off += (size_t)MROWS * INPUT * 2;    // 32 MB
    __bf16* WxP = (__bf16*)(ws + off); off += (size_t)INPUT * HIDDEN * 2;   //  4 MB
    __bf16* WhP = (__bf16*)(ws + off); off += (size_t)HIDDEN * HIDDEN * 2;  //  8 MB
    __bf16* hb0 = (__bf16*)(ws + off); off += (size_t)BATCH * HIDDEN * 2;   // 256 KB
    __bf16* hb1 = (__bf16*)(ws + off); off += (size_t)BATCH * HIDDEN * 2;   // 256 KB
    (void)ws_size; (void)in_sizes; (void)n_in; (void)out_size;

    // 1) precision/layout prep (parallel, bandwidth-trivial)
    cvt_f32_bf16_kernel<<<4096, 256, 0, stream>>>(x,  Xb,  (size_t)MROWS * INPUT);
    cvt_f32_bf16_kernel<<<64,   256, 0, stream>>>(h0, hb0, (size_t)BATCH * HIDDEN);
    pack_B_kernel<<<2048, 256, 0, stream>>>(Wx, WxP, INPUT,  HIDDEN);
    pack_B_kernel<<<4096, 256, 0, stream>>>(Wh, WhP, HIDDEN, HIDDEN);

    // 2) xproj = x @ Wx + bx straight into d_out slabs (LDS-staged B panel)
    gemm_xproj_kernel<<<dim3(HIDDEN / 64, MROWS / 256), 256, 128 * 1024, stream>>>(
        Xb, WxP, bx, out);

    // 3) 256 dependent steps; Wh (8MB bf16) stays L2-resident, h ping-pongs
    for (int t = 0; t < SEQ; ++t) {
        const __bf16* hc = (t & 1) ? hb1 : hb0;
        __bf16*       hn = (t & 1) ? hb0 : hb1;
        rnn_step_kernel<<<HIDDEN / 128, 256, 0, stream>>>(
            hc, WhP, bh, out + (size_t)t * BATCH * HIDDEN, hn);
    }

    // 4) hn = outputs[255]
    (void)hipMemcpyAsync(out + (size_t)SEQ * BATCH * HIDDEN,
                         out + (size_t)(SEQ - 1) * BATCH * HIDDEN,
                         (size_t)BATCH * HIDDEN * sizeof(float),
                         hipMemcpyDeviceToDevice, stream);
}
